// DepthSeparableConv2d_conv2_2_37572373905413
// MI455X (gfx1250) — matrix operand order repaired and checked by reading, not compiled, not run on hardware
//
#include <hip/hip_runtime.h>
#include <hip/hip_bf16.h>
#include <float.h>

#define HW    12544
#define WD    112
#define CIN   128
#define COUT  256
#define NBATCH 32
#define EPSV  1e-5f
#define INVN  (1.0f / 401408.0f)

typedef __attribute__((ext_vector_type(16))) __bf16 v16bf;
typedef __attribute__((ext_vector_type(8)))  float  v8f;

static __device__ __forceinline__ unsigned short f2bf(float f) {
    unsigned int u = __float_as_uint(f);
    u += 0x7FFFu + ((u >> 16) & 1u);     // round-to-nearest-even
    return (unsigned short)(u >> 16);
}

static __device__ __forceinline__ float wred_sum(float v) {
    for (int o = 16; o; o >>= 1) v += __shfl_xor(v, o, 32);
    return v;
}
static __device__ __forceinline__ float wred_max(float v) {
    for (int o = 16; o; o >>= 1) v = fmaxf(v, __shfl_xor(v, o, 32));
    return v;
}
static __device__ __forceinline__ float wred_min(float v) {
    for (int o = 16; o; o >>= 1) v = fminf(v, __shfl_xor(v, o, 32));
    return v;
}

// ---------------------------------------------------------------- init / weight convert
__global__ void k_init(const float* __restrict__ pw_w, unsigned short* __restrict__ wt,
                       float* sum1, float* sumsq1, float* pmax1, float* pmin1,
                       float* sum2, float* sumsq2, float* pmax2, float* pmin2) {
    int i = blockIdx.x * 256 + threadIdx.x;            // 0..32767
    if (i < COUT * CIN) wt[i] = f2bf(pw_w[i]);         // B matrix, [o][c] row-major
    if (i < CIN)  { sum1[i] = 0.f; sumsq1[i] = 0.f; }
    if (i < COUT) { sum2[i] = 0.f; sumsq2[i] = 0.f; }
    if (i < NBATCH * CIN)  { pmax1[i] = -FLT_MAX; pmin1[i] = FLT_MAX; }
    if (i < NBATCH * COUT) { pmax2[i] = -FLT_MAX; pmin2[i] = FLT_MAX; }
}

// ---------------------------------------------------------------- depthwise 3x3 + stats
__global__ void __launch_bounds__(256) k_dwconv(
        const float* __restrict__ x, const float* __restrict__ w9,
        const float* __restrict__ bias, float* __restrict__ y0,
        float* sum1, float* sumsq1, float* pmax1, float* pmin1) {
    const int p = blockIdx.x;                 // b*128 + c  (one block == one plane)
    const int c = p & (CIN - 1);
    const size_t base = (size_t)p * HW;
    float wv[9];
#pragma unroll
    for (int i = 0; i < 9; ++i) wv[i] = w9[c * 9 + i];
    const float bv = bias[c];
    float s = 0.f, q = 0.f, mx = -FLT_MAX, mn = FLT_MAX;
    for (int i = 0; i < 49; ++i) {
        int pix = i * 256 + threadIdx.x;
        int h = pix / WD, w = pix - h * WD;
        float acc = bv;
#pragma unroll
        for (int kh = 0; kh < 3; ++kh) {
            int ih = h + kh - 1;
            if (ih < 0 || ih >= WD) continue;
#pragma unroll
            for (int kw = 0; kw < 3; ++kw) {
                int iw = w + kw - 1;
                if (iw < 0 || iw >= WD) continue;
                acc = __builtin_fmaf(x[base + ih * WD + iw], wv[kh * 3 + kw], acc);
            }
        }
        y0[base + pix] = acc;
        s += acc; q = __builtin_fmaf(acc, acc, q);
        mx = fmaxf(mx, acc); mn = fminf(mn, acc);
    }
    __shared__ float red[4][8];
    int wave = threadIdx.x >> 5, lane = threadIdx.x & 31;
    s = wred_sum(s); q = wred_sum(q); mx = wred_max(mx); mn = wred_min(mn);
    if (lane == 0) { red[0][wave] = s; red[1][wave] = q; red[2][wave] = mx; red[3][wave] = mn; }
    __syncthreads();
    if (threadIdx.x == 0) {
        float S = 0.f, Q = 0.f, MX = -FLT_MAX, MN = FLT_MAX;
        for (int i = 0; i < 8; ++i) {
            S += red[0][i]; Q += red[1][i];
            MX = fmaxf(MX, red[2][i]); MN = fminf(MN, red[3][i]);
        }
        atomicAdd(&sum1[c], S); atomicAdd(&sumsq1[c], Q);
        pmax1[p] = MX; pmin1[p] = MN;         // sole writer of this plane
    }
}

// ---------------------------------------------------------------- BN fold + cut mask (1 block)
__global__ void k_bnmask(const float* __restrict__ sum, const float* __restrict__ sumsq,
                         const float* __restrict__ gamma, const float* __restrict__ beta,
                         const float* __restrict__ pmax, const float* __restrict__ pmin,
                         float* scale_c, float* shift_c,
                         float* scp, float* shp, int C, int planes, float thr) {
    for (int c = threadIdx.x; c < C; c += 256) {
        float mean = sum[c] * INVN;
        float var  = __builtin_fmaf(-mean, mean, sumsq[c] * INVN);
        float sc = gamma[c] * rsqrtf(var + EPSV);
        float sh = __builtin_fmaf(-mean, sc, beta[c]);
        scale_c[c] = sc; shift_c[c] = sh;
    }
    __syncthreads();
    for (int pidx = threadIdx.x; pidx < planes; pidx += 256) {
        int c = pidx % C;                       // p = b*C + c
        float sc = scale_c[c], sh = shift_c[c];
        float postmax = fmaxf(__builtin_fmaf(sc, pmax[pidx], sh),
                              __builtin_fmaf(sc, pmin[pidx], sh));
        bool keep = fmaxf(postmax, 0.f) >= thr; // max of relu'd plane vs threshold
        scp[pidx] = keep ? sc : 0.f;
        shp[pidx] = keep ? sh : 0.f;
    }
}

// ---------------------------------------------------------------- pointwise GEMM (bf16 WMMA)
// block: 256 thr = 8 waves; tile M=64 (one b, contiguous hw), N=256 (all), K=128.
// Double-buffered A staging (BN1+ReLU+cut fused into f32->bf16 convert); all 8 B
// fragments of a k-step live in distinct registers so the ds_load_b128s clause up
// and the 8 v_wmma issue back-to-back.
__global__ void __launch_bounds__(256) k_gemm(
        const float* __restrict__ y0, const unsigned short* __restrict__ wt,
        const float* __restrict__ pw_b,
        const float* __restrict__ scp, const float* __restrict__ shp,
        float* __restrict__ z) {
    __shared__ __align__(16) unsigned short Bs[COUT * 136];   // [n][k], 16B-aligned rows
    __shared__ __align__(16) unsigned short As[2][64 * 40];   // double-buffered [m][k32]
    const int tid = threadIdx.x;
    const int m0  = blockIdx.x * 64;
    const int b   = m0 / HW;
    const int hw0 = m0 - b * HW;                 // 64 | HW, tile stays in one image

    {   // stage all weights once: 32768 bf16 = 4096 uint4
        const uint4* src = (const uint4*)wt;
#pragma unroll
        for (int i = 0; i < 16; ++i) {
            int idx = i * 256 + tid;             // uint4 index, 16 per n-row
            int n = idx >> 4, k8 = idx & 15;
            *(uint4*)&Bs[n * 136 + k8 * 8] = src[idx];
        }
    }

    const int lane = tid & 31, l16 = lane & 15, hi = lane >> 4;
    const int wv   = tid >> 5;
    const int msub = wv & 3;                     // 4 m-subtiles of 16
    const int nb   = (wv >> 2) * 128;            // 2 wave-columns x 8 n-tiles

    // per-thread A-staging slots: idx = i*256+tid -> (channel co, row r)
    int coix[8], rix[8];
#pragma unroll
    for (int i = 0; i < 8; ++i) {
        int idx = i * 256 + tid;
        coix[i] = idx >> 6;                      // 0..31
        rix[i]  = idx & 63;
    }

    unsigned short st[8];
#pragma unroll
    for (int i = 0; i < 8; ++i) {                // prologue: fetch k-tile 0
        int pl = b * CIN + coix[i];
        float v = y0[(size_t)pl * HW + hw0 + rix[i]];
        st[i] = f2bf(fmaxf(__builtin_fmaf(scp[pl], v, shp[pl]), 0.f));
    }
#pragma unroll
    for (int i = 0; i < 8; ++i) As[0][rix[i] * 40 + coix[i]] = st[i];

    v8f acc[8] = {};
    union U { v16bf v; uint4 q[2]; };

    for (int kt = 0; kt < 4; ++kt) {
        const int kk = kt * 32;
        const int buf = kt & 1;
        __syncthreads();                         // As[buf] ready for everyone

        if (kt < 3) {                            // overlap next-tile HBM reads with WMMAs
#pragma unroll
            for (int i = 0; i < 8; ++i) {
                int pl = b * CIN + kk + 32 + coix[i];
                float v = y0[(size_t)pl * HW + hw0 + rix[i]];
                st[i] = f2bf(fmaxf(__builtin_fmaf(scp[pl], v, shp[pl]), 0.f));
            }
        }

        U af;                                    // lane row M=l16; hi half -> K 8-15/24-31
        const unsigned short* arow = &As[buf][(msub * 16 + l16) * 40 + hi * 8];
        af.q[0] = *(const uint4*)&arow[0];
        af.q[1] = *(const uint4*)&arow[16];

        U bfr[8];                                // lane col N=l16; hi half -> K 8-15/24-31
        const unsigned short* bbase = &Bs[(nb + l16) * 136 + kk + hi * 8];
#pragma unroll
        for (int nt = 0; nt < 8; ++nt) {
            bfr[nt].q[0] = *(const uint4*)&bbase[nt * 16 * 136];
            bfr[nt].q[1] = *(const uint4*)&bbase[nt * 16 * 136 + 16];
        }
#pragma unroll
        for (int nt = 0; nt < 8; ++nt)
            acc[nt] = __builtin_amdgcn_wmma_f32_16x16x32_bf16(
                false, af.v, false, bfr[nt].v, (short)0, acc[nt], false, false);

        if (kt < 3) {                            // fill the other buffer for next step
#pragma unroll
            for (int i = 0; i < 8; ++i) As[buf ^ 1][rix[i] * 40 + coix[i]] = st[i];
        }
    }

    const int hwp = hw0 + msub * 16 + hi * 8;    // C layout: vgpr r -> M = r + 8*hi
#pragma unroll
    for (int nt = 0; nt < 8; ++nt) {
        int n = nb + nt * 16 + l16;
        float bb = pw_b[n];
        size_t row = (size_t)(b * COUT + n) * HW + hwp;
        float4 lo = make_float4(acc[nt][0] + bb, acc[nt][1] + bb, acc[nt][2] + bb, acc[nt][3] + bb);
        float4 hi4 = make_float4(acc[nt][4] + bb, acc[nt][5] + bb, acc[nt][6] + bb, acc[nt][7] + bb);
        *(float4*)&z[row]     = lo;
        *(float4*)&z[row + 4] = hi4;
    }
}

// ---------------------------------------------------------------- BN2 stats over z
__global__ void __launch_bounds__(256) k_stats2(
        const float* __restrict__ z,
        float* sum2, float* sumsq2, float* pmax2, float* pmin2) {
    const int p = blockIdx.x;                    // b*256 + o
    const int c = p & (COUT - 1);
    const size_t base = (size_t)p * HW;
    float s = 0.f, q = 0.f, mx = -FLT_MAX, mn = FLT_MAX;
    for (int i = 0; i < 49; ++i) {
        float v = z[base + i * 256 + threadIdx.x];
        s += v; q = __builtin_fmaf(v, v, q);
        mx = fmaxf(mx, v); mn = fminf(mn, v);
    }
    __shared__ float red[4][8];
    int wave = threadIdx.x >> 5, lane = threadIdx.x & 31;
    s = wred_sum(s); q = wred_sum(q); mx = wred_max(mx); mn = wred_min(mn);
    if (lane == 0) { red[0][wave] = s; red[1][wave] = q; red[2][wave] = mx; red[3][wave] = mn; }
    __syncthreads();
    if (threadIdx.x == 0) {
        float S = 0.f, Q = 0.f, MX = -FLT_MAX, MN = FLT_MAX;
        for (int i = 0; i < 8; ++i) {
            S += red[0][i]; Q += red[1][i];
            MX = fmaxf(MX, red[2][i]); MN = fminf(MN, red[3][i]);
        }
        atomicAdd(&sum2[c], S); atomicAdd(&sumsq2[c], Q);
        pmax2[p] = MX; pmin2[p] = MN;
    }
}

// ---------------------------------------------------------------- BN2+ReLU+cut, in-place
__global__ void __launch_bounds__(256) k_apply2(
        float* __restrict__ z, const float* __restrict__ scp, const float* __restrict__ shp) {
    const int p = blockIdx.x;
    const float s = scp[p], h = shp[p];
    const size_t base = (size_t)p * HW;
    for (int i = 0; i < 49; ++i) {
        size_t idx = base + i * 256 + threadIdx.x;
        z[idx] = fmaxf(__builtin_fmaf(s, z[idx], h), 0.f);
    }
}

// ---------------------------------------------------------------- host
extern "C" void kernel_launch(void* const* d_in, const int* in_sizes, int n_in,
                              void* d_out, int out_size, void* d_ws, size_t ws_size,
                              hipStream_t stream) {
    const float* x      = (const float*)d_in[0];
    const float* dw_w   = (const float*)d_in[1];
    const float* dw_b   = (const float*)d_in[2];
    const float* bn1_g  = (const float*)d_in[3];
    const float* bn1_b  = (const float*)d_in[4];
    const float* pw_w   = (const float*)d_in[5];
    const float* pw_b   = (const float*)d_in[6];
    const float* bn2_g  = (const float*)d_in[7];
    const float* bn2_b  = (const float*)d_in[8];
    float* z = (float*)d_out;

    // workspace layout
    float* y0 = (float*)d_ws;                       // 51,380,224 f32
    float* f  = y0 + (size_t)NBATCH * CIN * HW;
    float* sum1   = f; f += CIN;
    float* sumsq1 = f; f += CIN;
    float* scale1 = f; f += CIN;
    float* shift1 = f; f += CIN;
    float* pmax1  = f; f += NBATCH * CIN;
    float* pmin1  = f; f += NBATCH * CIN;
    float* sc1p   = f; f += NBATCH * CIN;
    float* sh1p   = f; f += NBATCH * CIN;
    float* sum2   = f; f += COUT;
    float* sumsq2 = f; f += COUT;
    float* scale2 = f; f += COUT;
    float* shift2 = f; f += COUT;
    float* pmax2  = f; f += NBATCH * COUT;
    float* pmin2  = f; f += NBATCH * COUT;
    float* sc2p   = f; f += NBATCH * COUT;
    float* sh2p   = f; f += NBATCH * COUT;
    unsigned short* wt = (unsigned short*)f;        // 32768 bf16 (16B-aligned)

    k_init<<<128, 256, 0, stream>>>(pw_w, wt, sum1, sumsq1, pmax1, pmin1,
                                    sum2, sumsq2, pmax2, pmin2);
    k_dwconv<<<NBATCH * CIN, 256, 0, stream>>>(x, dw_w, dw_b, y0,
                                               sum1, sumsq1, pmax1, pmin1);
    k_bnmask<<<1, 256, 0, stream>>>(sum1, sumsq1, bn1_g, bn1_b, pmax1, pmin1,
                                    scale1, shift1, sc1p, sh1p,
                                    CIN, NBATCH * CIN, 4.0f);
    k_gemm<<<(NBATCH * HW) / 64, 256, 0, stream>>>(y0, wt, pw_b, sc1p, sh1p, z);
    k_stats2<<<NBATCH * COUT, 256, 0, stream>>>(z, sum2, sumsq2, pmax2, pmin2);
    k_bnmask<<<1, 256, 0, stream>>>(sum2, sumsq2, bn2_g, bn2_b, pmax2, pmin2,
                                    scale2, shift2, sc2p, sh2p,
                                    COUT, NBATCH * COUT, 0.001f);
    k_apply2<<<NBATCH * COUT, 256, 0, stream>>>(z, sc2p, sh2p);
    (void)in_sizes; (void)n_in; (void)out_size; (void)ws_size;
}
